// MultiLayerGIN_55198919688274
// MI455X (gfx1250) — compile-verified, matrix operand order checked by reading, not run on hardware
//
#include <hip/hip_runtime.h>
#include <hip/hip_bf16.h>

// ---------------------------------------------------------------------------
// MultiLayerGIN on MI455X (gfx1250, wave32, WMMA)
//  - scatter-add aggregation in f32 (L2-resident atomics; h+agg fit in 192MB L2)
//  - GEMMs via v_wmma_f32_16x16x32_bf16, f32 accumulate, bf16 operands
//  - 32x64 output tile per wave: 8 WMMAs per 6 fragment loads
//  - weights pre-transposed+converted to bf16 once per launch
// ---------------------------------------------------------------------------

typedef __bf16 bf16_t;
typedef __attribute__((ext_vector_type(16))) __bf16 v16bf;
typedef __attribute__((ext_vector_type(8)))  __bf16 v8bf;
typedef __attribute__((ext_vector_type(8)))  float  v8f;

#define NODES  50000
#define EDGES  800000
#define GRAPHS 256
#define IN_C   128
#define HID    256

// ---------------- elementwise helpers ----------------
__global__ void zero_f32_kernel(float* __restrict__ out, int n) {
  int i = blockIdx.x * blockDim.x + threadIdx.x;
  if (i < n) out[i] = 0.0f;
}

__global__ void copy_f32_kernel(const float* __restrict__ in, float* __restrict__ out, int n) {
  int i = blockIdx.x * blockDim.x + threadIdx.x;
  if (i < n) out[i] = in[i];
}

__global__ void f32_to_bf16_kernel(const float* __restrict__ in, bf16_t* __restrict__ out, int n) {
  int i = blockIdx.x * blockDim.x + threadIdx.x;
  if (i < n) out[i] = (bf16_t)in[i];
}

// w: K x HID row-major f32  ->  wt: HID x K row-major bf16 (i.e. w transposed)
__global__ void transpose_to_bf16_kernel(const float* __restrict__ w, bf16_t* __restrict__ wt, int K) {
  int i = blockIdx.x * blockDim.x + threadIdx.x;
  if (i < K * HID) {
    int n = i / K;     // output row = column of w
    int k = i % K;
    wt[i] = (bf16_t)w[(size_t)k * HID + n];
  }
}

// ---------------- scatter-add aggregation ----------------
// agg must be pre-seeded with h (fuses z = h + sum_{j->i} h[j])
__global__ void scatter_add_edges_kernel(const int* __restrict__ src, const int* __restrict__ dst,
                                         const float* __restrict__ h, float* __restrict__ agg,
                                         int nE, int H) {
  const int chunks = H >> 2;  // float4 chunks per row
  long long t = (long long)blockIdx.x * blockDim.x + threadIdx.x;
  if (t >= (long long)nE * chunks) return;
  int e = (int)(t / chunks);
  int c = ((int)(t % chunks)) << 2;
  int s = src[e];
  int d = dst[e];
  const float4 v = *(const float4*)(h + (size_t)s * H + c);
  float* p = agg + (size_t)d * H + c;
  atomicAdd(p + 0, v.x);
  atomicAdd(p + 1, v.y);
  atomicAdd(p + 2, v.z);
  atomicAdd(p + 3, v.w);
}

// ---------------- global add pool ----------------
__global__ void pool_add_kernel(const int* __restrict__ batch, const float* __restrict__ h,
                                float* __restrict__ out) {
  long long t = (long long)blockIdx.x * blockDim.x + threadIdx.x;
  if (t >= (long long)NODES * (HID / 4)) return;
  int n = (int)(t / (HID / 4));
  int c = ((int)(t % (HID / 4))) << 2;
  int g = batch[n];
  const float4 v = *(const float4*)(h + (size_t)n * HID + c);
  float* p = out + (size_t)g * HID + c;
  atomicAdd(p + 0, v.x);
  atomicAdd(p + 1, v.y);
  atomicAdd(p + 2, v.z);
  atomicAdd(p + 3, v.w);
}

// ---------------- WMMA GEMM ----------------
union FragU { v16bf v; v8bf h[2]; };

// Load a 16x32 bf16 fragment (A layout; B^T uses the same layout since we store
// weights transposed). ISA lane map: lanes 0-15 hold K {0..7,16..23}, lanes
// 16-31 hold K {8..15,24..31} of their row.
__device__ __forceinline__ v16bf load_frag16x32(const bf16_t* __restrict__ base,
                                                int stride, int rowMax,
                                                int r0, int k0, int lane) {
  int half = lane >> 4;
  int m    = lane & 15;
  int r    = r0 + m;
  if (r > rowMax) r = rowMax;  // clamp (OOB rows never stored)
  const bf16_t* p = base + (size_t)r * stride + k0 + half * 8;
  FragU f;
  f.h[0] = *(const v8bf*)(p);
  f.h[1] = *(const v8bf*)(p + 16);
  return f.v;
}

// C = relu(A @ Wt^T + bias). A: M x K bf16 row-major. Wt: HID x K bf16 (B^T).
// 256 threads = 8 waves; each wave computes a 32 x 64 output tile
// (2 row-groups x 4 col-tiles = 8 WMMAs per 6 fragment loads per k-step).
__global__ void __launch_bounds__(256)
gin_gemm_wmma_kernel(const bf16_t* __restrict__ A,
                     const bf16_t* __restrict__ Wt,
                     const float*  __restrict__ bias,
                     int M, int K,
                     float*  __restrict__ outF,   // optional f32 out (M x HID)
                     bf16_t* __restrict__ outB) { // optional bf16 out (M x HID)
  const int lane = threadIdx.x & 31;
  const int wave = threadIdx.x >> 5;
  const int r0   = blockIdx.x * 256 + wave * 32;  // 32 rows per wave
  const int c0   = blockIdx.y * 64;               // 64 cols per block
  if (r0 >= M) return;  // wave-uniform: EXEC stays all-ones for WMMA

  v8f acc[2][4] = {};
  for (int k0 = 0; k0 < K; k0 += 32) {
    v16bf a0 = load_frag16x32(A,  K, M - 1,   r0,      k0, lane);
    v16bf a1 = load_frag16x32(A,  K, M - 1,   r0 + 16, k0, lane);
    v16bf b0 = load_frag16x32(Wt, K, HID - 1, c0 +  0, k0, lane);
    v16bf b1 = load_frag16x32(Wt, K, HID - 1, c0 + 16, k0, lane);
    v16bf b2 = load_frag16x32(Wt, K, HID - 1, c0 + 32, k0, lane);
    v16bf b3 = load_frag16x32(Wt, K, HID - 1, c0 + 48, k0, lane);
    acc[0][0] = __builtin_amdgcn_wmma_f32_16x16x32_bf16(false, a0, false, b0, (short)0, acc[0][0], false, false);
    acc[0][1] = __builtin_amdgcn_wmma_f32_16x16x32_bf16(false, a0, false, b1, (short)0, acc[0][1], false, false);
    acc[0][2] = __builtin_amdgcn_wmma_f32_16x16x32_bf16(false, a0, false, b2, (short)0, acc[0][2], false, false);
    acc[0][3] = __builtin_amdgcn_wmma_f32_16x16x32_bf16(false, a0, false, b3, (short)0, acc[0][3], false, false);
    acc[1][0] = __builtin_amdgcn_wmma_f32_16x16x32_bf16(false, a1, false, b0, (short)0, acc[1][0], false, false);
    acc[1][1] = __builtin_amdgcn_wmma_f32_16x16x32_bf16(false, a1, false, b1, (short)0, acc[1][1], false, false);
    acc[1][2] = __builtin_amdgcn_wmma_f32_16x16x32_bf16(false, a1, false, b2, (short)0, acc[1][2], false, false);
    acc[1][3] = __builtin_amdgcn_wmma_f32_16x16x32_bf16(false, a1, false, b3, (short)0, acc[1][3], false, false);
  }

  // C/D layout: lanes 0-15: VGPR v -> (M=v, N=lane); lanes 16-31: (M=v+8, N=lane-16)
  const int half = lane >> 4;
  const int ncol = lane & 15;
#pragma unroll
  for (int g = 0; g < 2; ++g) {
    int rbase = r0 + g * 16;
#pragma unroll
    for (int t = 0; t < 4; ++t) {
      int col  = c0 + t * 16 + ncol;
      float bv = bias[col];
#pragma unroll
      for (int v = 0; v < 8; ++v) {
        int r = rbase + v + half * 8;
        if (r < M) {
          float val = acc[g][t][v] + bv;
          val = val > 0.0f ? val : 0.0f;
          if (outF) outF[(size_t)r * HID + col] = val;
          if (outB) outB[(size_t)r * HID + col] = (bf16_t)val;
        }
      }
    }
  }
}

// ---------------- host-side layer driver ----------------
static void run_gin_layer(const float* h_in, int Hin,
                          const bf16_t* w1t, const float* b1,
                          const bf16_t* w2t, const float* b2,
                          const int* src, const int* dst,
                          float* agg, bf16_t* zb, bf16_t* yb, float* h_out,
                          hipStream_t stream) {
  const int nIn = NODES * Hin;
  // agg = h  (seed so scatter-add yields h + sum of neighbors)
  copy_f32_kernel<<<(nIn + 255) / 256, 256, 0, stream>>>(h_in, agg, nIn);
  // agg[dst] += h[src]
  long long eThreads = (long long)EDGES * (Hin >> 2);
  scatter_add_edges_kernel<<<(int)((eThreads + 255) / 256), 256, 0, stream>>>(
      src, dst, h_in, agg, EDGES, Hin);
  // zb = bf16(agg)
  f32_to_bf16_kernel<<<(nIn + 255) / 256, 256, 0, stream>>>(agg, zb, nIn);
  // yb = relu(zb @ w1 + b1)   (bf16 out, feeds next GEMM)
  dim3 grid((NODES + 255) / 256, HID / 64);
  gin_gemm_wmma_kernel<<<grid, 256, 0, stream>>>(zb, w1t, b1, NODES, Hin, nullptr, yb);
  // h_out = relu(yb @ w2 + b2)  (f32 out, feeds next layer / pooling)
  gin_gemm_wmma_kernel<<<grid, 256, 0, stream>>>(yb, w2t, b2, NODES, HID, h_out, nullptr);
}

extern "C" void kernel_launch(void* const* d_in, const int* in_sizes, int n_in,
                              void* d_out, int out_size, void* d_ws, size_t ws_size,
                              hipStream_t stream) {
  const float* x    = (const float*)d_in[0];
  const int*   edge = (const int*)d_in[1];   // [2, E]: src row then dst row
  const int*   batch= (const int*)d_in[2];
  const float* w1_0 = (const float*)d_in[3];
  const float* b1_0 = (const float*)d_in[4];
  const float* w2_0 = (const float*)d_in[5];
  const float* b2_0 = (const float*)d_in[6];
  const float* w1s  = (const float*)d_in[7]; // [3, 256, 256]
  const float* b1s  = (const float*)d_in[8]; // [3, 256]
  const float* w2s  = (const float*)d_in[9];
  const float* b2s  = (const float*)d_in[10];
  const int* src = edge;
  const int* dst = edge + EDGES;
  float* out = (float*)d_out;

  // ---- workspace carve-up (~206 MB) ----
  char* ws = (char*)d_ws;
  float* agg = (float*)ws;  ws += (size_t)NODES * HID * sizeof(float);
  float* h_a = (float*)ws;  ws += (size_t)NODES * HID * sizeof(float);
  float* h_b = (float*)ws;  ws += (size_t)NODES * HID * sizeof(float);
  bf16_t* zb = (bf16_t*)ws; ws += (size_t)NODES * HID * sizeof(bf16_t);
  bf16_t* yb = (bf16_t*)ws; ws += (size_t)NODES * HID * sizeof(bf16_t);
  bf16_t* w1t0 = (bf16_t*)ws; ws += (size_t)HID * IN_C * sizeof(bf16_t);
  bf16_t* w2t0 = (bf16_t*)ws; ws += (size_t)HID * HID * sizeof(bf16_t);
  bf16_t* w1t[3], *w2t[3];
  for (int i = 0; i < 3; ++i) { w1t[i] = (bf16_t*)ws; ws += (size_t)HID * HID * sizeof(bf16_t); }
  for (int i = 0; i < 3; ++i) { w2t[i] = (bf16_t*)ws; ws += (size_t)HID * HID * sizeof(bf16_t); }
  (void)ws_size; (void)in_sizes; (void)n_in;

  // ---- weight prep: transpose + convert to bf16 (tiny, once per launch) ----
  transpose_to_bf16_kernel<<<(HID * IN_C + 255) / 256, 256, 0, stream>>>(w1_0, w1t0, IN_C);
  transpose_to_bf16_kernel<<<(HID * HID + 255) / 256, 256, 0, stream>>>(w2_0, w2t0, HID);
  for (int i = 0; i < 3; ++i) {
    transpose_to_bf16_kernel<<<(HID * HID + 255) / 256, 256, 0, stream>>>(
        w1s + (size_t)i * HID * HID, w1t[i], HID);
    transpose_to_bf16_kernel<<<(HID * HID + 255) / 256, 256, 0, stream>>>(
        w2s + (size_t)i * HID * HID, w2t[i], HID);
  }

  // ---- 4 GIN layers (ping-pong h between h_a / h_b) ----
  run_gin_layer(x,   IN_C, w1t0,   b1_0,            w2t0,   b2_0,            src, dst, agg, zb, yb, h_a, stream);
  run_gin_layer(h_a, HID,  w1t[0], b1s + 0 * HID,   w2t[0], b2s + 0 * HID,   src, dst, agg, zb, yb, h_b, stream);
  run_gin_layer(h_b, HID,  w1t[1], b1s + 1 * HID,   w2t[1], b2s + 1 * HID,   src, dst, agg, zb, yb, h_a, stream);
  run_gin_layer(h_a, HID,  w1t[2], b1s + 2 * HID,   w2t[2], b2s + 2 * HID,   src, dst, agg, zb, yb, h_b, stream);

  // ---- global add pool -> out [G, HID] ----
  zero_f32_kernel<<<(out_size + 255) / 256, 256, 0, stream>>>(out, out_size);
  long long pThreads = (long long)NODES * (HID / 4);
  pool_add_kernel<<<(int)((pThreads + 255) / 256), 256, 0, stream>>>(batch, h_b, out);
}
